// UNet_63110249447457
// MI455X (gfx1250) — matrix operand order repaired and checked by reading, not compiled, not run on hardware
//
#include <hip/hip_runtime.h>
#include <hip/hip_bf16.h>

typedef __attribute__((ext_vector_type(16))) _Float16 v16h;
typedef __attribute__((ext_vector_type(8)))  _Float16 v8h;
typedef __attribute__((ext_vector_type(8)))  float    v8f;

// ---------------- problem dims ----------------
static constexpr int Z0c=16, Y0c=96, X0c=96, V0c=Z0c*Y0c*X0c;   // 147456
static constexpr int Z1c=8,  Y1c=48, X1c=48, V1c=Z1c*Y1c*X1c;   // 18432
static constexpr int Z2c=4,  Y2c=24, X2c=24, V2c=Z2c*Y2c*X2c;   // 2304
static constexpr int Z3c=1,  Y3c=12, X3c=12, V3c=Z3c*Y3c*X3c;   // 144
static constexpr int NHc=12000;                                 // history voxels

static inline int cdiv(int a, int b) { return (a + b - 1) / b; }

// ---------------- weight packing into WMMA B-lane order ----------------
// Reduction index k = tap*Cin + cin ; B tile block (kb,nb) is 512 contiguous
// halves; lane L's 16 halves contiguous: element (K = kb*32 + (h&7) + 16*(h>>3) + 8*(L>>4),
// N = nb*16 + (L&15)).  Source W is (Cout, Cin, nTaps) row-major (OIDHW).
__global__ void pack_w(const float* __restrict__ W, _Float16* __restrict__ dst,
                       int Cout, int Cin, int nTaps)
{
  int NB = (Cout + 15) >> 4;
  int K  = nTaps * Cin;
  int KB = (K + 31) >> 5;
  size_t total = (size_t)KB * NB * 512;
  size_t idx = (size_t)blockIdx.x * blockDim.x + threadIdx.x;
  if (idx >= total) return;
  int e = (int)(idx & 511);
  size_t blk = idx >> 9;
  int nb = (int)(blk % NB);
  int kb = (int)(blk / NB);
  int L = e >> 4, h = e & 15;
  int k = (kb << 5) + (h & 7) + ((h >> 3) << 4) + ((L >> 4) << 3);
  int n = (nb << 4) + (L & 15);
  float v = 0.f;
  if (k < K && n < Cout) {
    int tap = k / Cin, cin = k - tap * Cin;
    v = W[((size_t)n * Cin + cin) * nTaps + tap];
  }
  dst[idx] = (_Float16)v;
}

// ---------------- generic implicit-GEMM conv / linear via WMMA ----------------
// Fully shape-specialized: all divisions fold to mul-shift, transposed /
// dilated paths are dead-code eliminated per instantiation, KB/NB constant.
// 64 threads = 2 waves; block = one 16-voxel M tile x all COUT.
// Wave w covers n-tiles nb = w, w+2, ... (max 8 tiles -> COUT <= 256).
// Normal conv:      in = z*S + j - P            (U == 1)
// Transposed conv:  in = (z + P - j)/U if divisible (S == 1, U == 2)
// A staging: thread stages 8 CONSECUTIVE k for one matrix row
//   (L = tid>>1, k = kb*32 + q + ((tid&1)<<4) + (wave<<3)),
// so (tap,jz,jy,jx) advance incrementally and 8 halves = one 16B ds_store.
template<int CIN, int COUT, int TAPS,
         int IZ, int IY, int IX, int OZ, int OY, int OX,
         int S, int U, int TR, int PZ, int PY, int PX>
__global__ __launch_bounds__(64)
void conv_wmma(const float* __restrict__ in, const _Float16* __restrict__ wgt,
               const float* __restrict__ gam, const float* __restrict__ beta,
               const float* __restrict__ mask, const float* __restrict__ res,
               float* __restrict__ out,
               int inPitch, int inOff, int outPitch, int outOff,
               int resPitch, int resOff, int act, int useMask)
{
  constexpr int OYX  = OY * OX;
  constexpr int OVOX = OZ * OYX;
  constexpr int K    = TAPS * CIN;
  constexpr int KB   = (K + 31) / 32;
  constexpr int NB   = (COUT + 15) / 16;

  __shared__ __align__(32) _Float16 Alds[512];
  const int tid  = threadIdx.x;
  const int lane = tid & 31;
  const int wave = tid >> 5;
  const int tile = blockIdx.x;
  const int myNB = (NB - wave + 1) >> 1;   // n-tiles handled by this wave

  // ---- per-thread staging invariants ----
  const int m    = (tid >> 1) & 15;              // matrix row staged by this thread
  const int kgrp = ((tid & 1) << 4) + ((tid >> 5) << 3);
  const int vo_s = tile * 16 + m;
  const bool mvalid = vo_s < OVOX;
  int z = 0, y = 0, x = 0;
  if (mvalid) {
    z = vo_s / OYX;                              // constant divisor -> mul/shift
    int rr = vo_s - z * OYX;
    y = rr / OX;
    x = rr - y * OX;
  }

  auto mkaddr = [&](int jz, int jy, int jx, bool& okk) -> size_t {
    int nz = TR ? (z + PZ - jz) : (z * S + jz - PZ);
    int ny = TR ? (y + PY - jy) : (y * S + jy - PY);
    int nx = TR ? (x + PX - jx) : (x * S + jx - PX);
    bool o = ((nz | ny | nx) >= 0);
    if (U == 2) { o = o && (((nz | ny | nx) & 1) == 0); nz >>= 1; ny >>= 1; nx >>= 1; }
    okk = o && (nz < IZ) && (ny < IY) && (nx < IX);
    return (size_t)((nz * IY + ny) * IX + nx) * inPitch + inOff;
  };

  v8f acc[8];
#pragma unroll
  for (int i = 0; i < 8; i++)
#pragma unroll
    for (int j = 0; j < 8; j++) acc[i][j] = 0.f;

  for (int kb = 0; kb < KB; ++kb) {
    __syncthreads();                       // previous A fully consumed

    // ---- incremental (tap, cin) walk over 8 consecutive k ----
    int k0  = (kb << 5) + kgrp;
    int tap = k0 / CIN;                    // constant divisor
    int cin = k0 - tap * CIN;
    int jz = 0, jy = 0, jx = 0;
    bool ok = false;
    size_t sbase = 0;
    if (mvalid && tap < TAPS) {
      jz = tap / 9; int jr = tap - jz * 9; jy = jr / 3; jx = jr - jy * 3;
      sbase = mkaddr(jz, jy, jx, ok);
    }
    v8h st;
#pragma unroll
    for (int q = 0; q < 8; q++) {
      float v = 0.f;
      if (ok) v = in[sbase + cin];         // contiguous cin -> coalesced
      st[q] = (_Float16)v;
      if (++cin == CIN) {                  // tap carry (rare)
        cin = 0; ++tap;
        ++jx; if (jx == 3) { jx = 0; ++jy; if (jy == 3) { jy = 0; ++jz; } }
        ok = false;
        if (mvalid && tap < TAPS) sbase = mkaddr(jz, jy, jx, ok);
      }
    }
    *(v8h*)(&Alds[tid << 3]) = st;         // single 16B ds_store
    __syncthreads();

    v16h a = *(const v16h*)(&Alds[lane << 4]);
    const _Float16* wb = wgt + (size_t)kb * NB * 512 + (lane << 4);
    if (kb + 1 < KB)                       // stream next weight chunk
      __builtin_prefetch(wb + (size_t)NB * 512, 0, 0);
#pragma unroll
    for (int i = 0; i < 8; i++) {
      if (i < myNB) {
        int nb = wave + (i << 1);
        v16h b = *(const v16h*)(wb + (size_t)nb * 512);
        acc[i] = __builtin_amdgcn_wmma_f32_16x16x32_f16(
            false, a, false, b, (short)0, acc[i], false, false);
      }
    }
  }

  // ---- epilogue: BN scale/bias, residual, act, mask ----
  const int group = lane >> 4;
  const int nl = lane & 15;
#pragma unroll
  for (int i = 0; i < 8; i++) {
    if (i < myNB) {
      int nb = wave + (i << 1);
      int n = nb * 16 + nl;
      float sc = 1.f, bs = 0.f;
      if (n < COUT) {
        if (gam)  sc = gam[n] * rsqrtf(1.0f + 1.0e-3f);
        if (beta) bs = beta[n];
      }
#pragma unroll
      for (int r = 0; r < 8; r++) {
        int mm = r + (group << 3);
        int vo = tile * 16 + mm;
        if (vo < OVOX && n < COUT) {
          float val = acc[i][r] * sc + bs;
          if (res) val += res[(size_t)vo * resPitch + resOff + n];
          if (act == 1) val = fmaxf(val, 0.f);
          else if (act == 2) val = 0.5f * val * (1.f + erff(val * 0.70710678f));
          if (useMask) val *= mask[vo];
          out[(size_t)vo * outPitch + outOff + n] = val;
        }
      }
    }
  }
}

// ---------------- host-side launcher ----------------
template<int CIN, int COUT, int TAPS,
         int IZ, int IY, int IX, int OZ, int OY, int OX,
         int S, int U, int TR, int PZ, int PY, int PX>
static void conv(hipStream_t st, const float* in, const _Float16* w,
                 const float* g, const float* b, const float* mask, const float* res,
                 float* out, int inPitch, int inOff, int outPitch, int outOff,
                 int resPitch, int resOff, int act, int useMask)
{
  constexpr int oV = OZ * OY * OX;
  conv_wmma<CIN,COUT,TAPS,IZ,IY,IX,OZ,OY,OX,S,U,TR,PZ,PY,PX>
      <<<cdiv(oV, 16), 64, 0, st>>>(in, w, g, b, mask, res, out,
      inPitch, inOff, outPitch, outOff, resPitch, resOff, act, useMask);
}

// ---------------- elementwise helpers ----------------
__global__ void scatter_in(const float* __restrict__ vf, const int* __restrict__ vc,
                           float* __restrict__ dense, float* __restrict__ m0, int N)
{
  int idx = blockIdx.x * blockDim.x + threadIdx.x;
  if (idx >= N * 38) return;
  int i = idx / 38, c = idx - i * 38;
  int z = vc[i * 4 + 1], y = vc[i * 4 + 2], x = vc[i * 4 + 3];
  int vo = (z * Y0c + y) * X0c + x;
  dense[(size_t)vo * 38 + c] = vf[idx];
  if (c == 0) m0[vo] = 1.f;
}

__global__ void down_mask_k(const float* __restrict__ mi, float* __restrict__ mo,
                            int iZ, int iY, int iX, int oZ, int oY, int oX,
                            int pz, int py, int px)
{
  int vo = blockIdx.x * blockDim.x + threadIdx.x;
  if (vo >= oZ * oY * oX) return;
  int z = vo / (oY * oX), rr = vo % (oY * oX), y = rr / oX, x = rr % oX;
  float any = 0.f;
  for (int jz = 0; jz < 3; jz++)
    for (int jy = 0; jy < 3; jy++)
      for (int jx = 0; jx < 3; jx++) {
        int nz = 2 * z + jz - pz, ny = 2 * y + jy - py, nx = 2 * x + jx - px;
        if (nz >= 0 && nz < iZ && ny >= 0 && ny < iY && nx >= 0 && nx < iX)
          if (mi[(nz * iY + ny) * iX + nx] > 0.f) any = 1.f;
      }
  mo[vo] = any;
}

// t = xm + xc[:,2c] + xc[:,2c+1]   (xc has pitch 2C)
__global__ void pair_sum_k(const float* __restrict__ xm, const float* __restrict__ xc,
                           float* __restrict__ t, int V, int C)
{
  int idx = blockIdx.x * blockDim.x + threadIdx.x;
  if (idx >= V * C) return;
  int v = idx / C, c = idx - v * C;
  t[idx] = xm[idx] + xc[(size_t)v * 2 * C + 2 * c] + xc[(size_t)v * 2 * C + 2 * c + 1];
}

__global__ void centers_k(float* __restrict__ cent, float* __restrict__ pc, int V)
{
  int v = blockIdx.x * blockDim.x + threadIdx.x;
  if (v >= V) return;
  int z = v / (Y1c * X1c), rr = v % (Y1c * X1c), y = rr / X1c, x = rr % X1c;
  float cx = (x + 0.5f) * 0.4f - 9.6f;
  float cy = (y + 0.5f) * 0.4f - 9.6f;
  float cz = (z + 0.5f) * 0.8f - 3.2f;
  cent[v * 3 + 0] = cx; cent[v * 3 + 1] = cy; cent[v * 3 + 2] = cz;
  pc[v * 4 + 0] = 0.f; pc[v * 4 + 1] = cx; pc[v * 4 + 2] = cy; pc[v * 4 + 3] = cz;
}

__global__ void hist_scatter_k(const float* __restrict__ hvf, const int* __restrict__ hvc,
                               const float* __restrict__ hm, const float* __restrict__ hmlp,
                               const float* __restrict__ te0,
                               float* __restrict__ hd, float* __restrict__ mh, int NH)
{
  int idx = blockIdx.x * blockDim.x + threadIdx.x;
  if (idx >= NH * 64) return;
  int i = idx >> 6, c = idx & 63;
  int z = hvc[i * 4 + 1], y = hvc[i * 4 + 2], x = hvc[i * 4 + 3];
  int vo = (z * Y1c + y) * X1c + x;
  float v = (hvf[idx] + te0[c] + hmlp[idx]) * hm[i];
  hd[(size_t)vo * 64 + c] = v;          // coords unique -> plain store
  if (c == 0) mh[vo] = 1.f;
}

// fin = xu3 + (te1 + pe)*m1 + hd ;  mu = max(m1, mh)
__global__ void fuse_k(const float* __restrict__ xu3, const float* __restrict__ pe,
                       const float* __restrict__ te1, const float* __restrict__ m1,
                       const float* __restrict__ hd, const float* __restrict__ mh,
                       float* __restrict__ fin, float* __restrict__ mu, int V)
{
  int idx = blockIdx.x * blockDim.x + threadIdx.x;
  if (idx >= V * 64) return;
  int v = idx >> 6, c = idx & 63;
  fin[idx] = xu3[idx] + (te1[c] + pe[idx]) * m1[v] + hd[idx];
  if (c == 0) mu[v] = fmaxf(m1[v], mh[v]);
}

__global__ void copy_mask_k(const float* __restrict__ mu, float* __restrict__ o, int V)
{
  int v = blockIdx.x * blockDim.x + threadIdx.x;
  if (v < V) o[v] = mu[v];
}

// d_in layout assumption: inputs 0..6 in setup_inputs() insertion order, then
// params flattened as a jax pytree (dict keys sorted, lists in order):
//  7-9 conv1{W,b,g}  10-18 conv2[0..2]{W,b,g}  19-27 conv3  28-36 conv4
// 37-39 conv_input  40-48 fusion  49-52 gauss{W1,W2,b1,b2}  53-55 inv3
// 56-58 inv4  59-67 out_conv  68-71 pos_emb{W1,W2,b1,b2}  72 time_emb
// 73-75 up_m3  76-78 up_m4  79-84 up_t3{W1,W2,b1,b2,g1,g2}  85-90 up_t4
extern "C" void kernel_launch(void* const* d_in, const int* in_sizes, int n_in,
                              void* d_out, int out_size, void* d_ws, size_t ws_size,
                              hipStream_t stream)
{
  (void)n_in; (void)out_size; (void)ws_size;
  const float* vfeat = (const float*)d_in[0];
  const int*   vcrd  = (const int*)d_in[1];
  const float* hfeat = (const float*)d_in[3];
  const int*   hcrd  = (const int*)d_in[4];
  const float* hpos  = (const float*)d_in[5];
  const float* hmask = (const float*)d_in[6];
  const int N  = in_sizes[0] / 38;
  const int NH = in_sizes[3] / 64;   // == NHc
#define P(i) ((const float*)d_in[i])

  char* base = (char*)d_ws;
  size_t off = 0;
  auto alloc  = [&](size_t bytes) -> void* {
    void* p = base + off; off = (off + bytes + 255) & ~(size_t)255; return p; };
  auto allocF = [&](size_t n) -> float* { return (float*)alloc(n * 4); };

  auto packw = [&](int idx, int Co, int Ci, int t) -> _Float16* {
    size_t total = (size_t)((t * Ci + 31) / 32) * ((Co + 15) / 16) * 512;
    _Float16* p = (_Float16*)alloc(total * 2);
    pack_w<<<(unsigned)((total + 255) / 256), 256, 0, stream>>>(P(idx), p, Co, Ci, t);
    return p;
  };

  _Float16* w_ci  = packw(37, 32, 38, 27);
  _Float16* w_c1  = packw(7,  32, 32, 27);
  _Float16* w_20  = packw(10, 64, 32, 27);
  _Float16* w_21  = packw(13, 64, 64, 27);
  _Float16* w_22  = packw(16, 64, 64, 27);
  _Float16* w_30  = packw(19, 128, 64, 27);
  _Float16* w_31  = packw(22, 128, 128, 27);
  _Float16* w_32  = packw(25, 128, 128, 27);
  _Float16* w_40  = packw(28, 256, 128, 27);
  _Float16* w_41  = packw(31, 256, 256, 27);
  _Float16* w_42  = packw(34, 256, 256, 27);
  _Float16* w_t4a = packw(85, 256, 256, 27);
  _Float16* w_t4b = packw(86, 256, 256, 27);
  _Float16* w_m4  = packw(76, 256, 512, 27);
  _Float16* w_i4  = packw(56, 128, 256, 27);
  _Float16* w_t3a = packw(79, 128, 128, 27);
  _Float16* w_t3b = packw(80, 128, 128, 27);
  _Float16* w_m3  = packw(73, 128, 256, 27);
  _Float16* w_i3  = packw(53, 64, 128, 27);
  _Float16* w_f0  = packw(40, 64, 64, 27);
  _Float16* w_f1  = packw(43, 64, 64, 27);
  _Float16* w_f2  = packw(46, 64, 64, 27);
  _Float16* w_o0  = packw(59, 64, 64, 27);
  _Float16* w_o1  = packw(62, 64, 64, 27);
  _Float16* w_o2  = packw(65, 64, 64, 27);
  _Float16* w_pe1 = packw(68, 64, 3, 1);
  _Float16* w_pe2 = packw(69, 64, 64, 1);
  _Float16* w_g1  = packw(49, 64, 64, 1);
  _Float16* w_g2  = packw(50, 30, 64, 1);

  float* dense = allocF((size_t)V0c * 38);
  float* m0 = allocF(V0c); float* m1 = allocF(V1c);
  float* m2 = allocF(V2c); float* m3 = allocF(V3c);
  float* mh = allocF(V1c); float* mu = allocF(V1c);
  float* b0a = allocF((size_t)V0c * 32);
  float* b0b = allocF((size_t)V0c * 32);
  float* l1a = allocF((size_t)V1c * 64);
  float* l1b = allocF((size_t)V1c * 64);
  float* peh = allocF((size_t)V1c * 64);
  float* pe  = allocF((size_t)V1c * 64);
  float* hd  = allocF((size_t)V1c * 64);
  float* cent= allocF((size_t)V1c * 3);
  float* hh  = allocF((size_t)NHc * 64);
  float* hm2 = allocF((size_t)NHc * 64);
  float* l2a = allocF((size_t)V2c * 128);
  float* l2b = allocF((size_t)V2c * 128);
  float* xu4 = allocF((size_t)V2c * 128);
  float* h2b = allocF((size_t)V2c * 128);
  float* xm3 = allocF((size_t)V2c * 128);
  float* t3  = allocF((size_t)V2c * 128);
  float* xc2 = allocF((size_t)V2c * 256);
  float* l3a = allocF((size_t)V3c * 256);
  float* l3b = allocF((size_t)V3c * 256);
  float* h3b = allocF((size_t)V3c * 256);
  float* xm4 = allocF((size_t)V3c * 256);
  float* t4  = allocF((size_t)V3c * 256);
  float* xc4 = allocF((size_t)V3c * 512);

  float* out_g  = (float*)d_out;
  float* out_pc = out_g + (size_t)V1c * 30;
  float* out_sv = out_pc + (size_t)V1c * 4;
  float* out_mk = out_sv + (size_t)V1c * 64;

  hipMemsetAsync(dense, 0, (size_t)V0c * 38 * 4, stream);
  hipMemsetAsync(m0, 0, (size_t)V0c * 4, stream);
  hipMemsetAsync(hd, 0, (size_t)V1c * 64 * 4, stream);
  hipMemsetAsync(mh, 0, (size_t)V1c * 4, stream);

  scatter_in<<<cdiv(N * 38, 256), 256, 0, stream>>>(vfeat, vcrd, dense, m0, N);
  down_mask_k<<<cdiv(V1c, 256), 256, 0, stream>>>(m0, m1, Z0c,Y0c,X0c, Z1c,Y1c,X1c, 1,1,1);
  down_mask_k<<<cdiv(V2c, 256), 256, 0, stream>>>(m1, m2, Z1c,Y1c,X1c, Z2c,Y2c,X2c, 1,1,1);
  down_mask_k<<<cdiv(V3c, 256), 256, 0, stream>>>(m2, m3, Z2c,Y2c,X2c, Z3c,Y3c,X3c, 0,1,1);

  // encoder
  conv<38,32,27,  Z0c,Y0c,X0c, Z0c,Y0c,X0c, 1,1,0, 1,1,1>(stream, dense, w_ci, P(39), P(38), m0, nullptr, b0a, 38,0, 32,0, 0,0, 1,1);
  conv<32,32,27,  Z0c,Y0c,X0c, Z0c,Y0c,X0c, 1,1,0, 1,1,1>(stream, b0a, w_c1, P(9), P(8), m0, nullptr, b0b, 32,0, 32,0, 0,0, 1,1);
  conv<32,64,27,  Z0c,Y0c,X0c, Z1c,Y1c,X1c, 2,1,0, 1,1,1>(stream, b0b, w_20, P(12), P(11), m1, nullptr, l1a, 32,0, 64,0, 0,0, 1,1);
  conv<64,64,27,  Z1c,Y1c,X1c, Z1c,Y1c,X1c, 1,1,0, 1,1,1>(stream, l1a, w_21, P(15), P(14), m1, nullptr, l1b, 64,0, 64,0, 0,0, 1,1);
  conv<64,64,27,  Z1c,Y1c,X1c, Z1c,Y1c,X1c, 1,1,0, 1,1,1>(stream, l1b, w_22, P(18), P(17), m1, nullptr, l1a, 64,0, 64,0, 0,0, 1,1);
  conv<64,128,27, Z1c,Y1c,X1c, Z2c,Y2c,X2c, 2,1,0, 1,1,1>(stream, l1a, w_30, P(21), P(20), m2, nullptr, l2a, 64,0, 128,0, 0,0, 1,1);
  conv<128,128,27,Z2c,Y2c,X2c, Z2c,Y2c,X2c, 1,1,0, 1,1,1>(stream, l2a, w_31, P(24), P(23), m2, nullptr, l2b, 128,0, 128,0, 0,0, 1,1);
  conv<128,128,27,Z2c,Y2c,X2c, Z2c,Y2c,X2c, 1,1,0, 1,1,1>(stream, l2b, w_32, P(27), P(26), m2, nullptr, xc2, 128,0, 256,0, 0,0, 1,1); // c3 -> xc2[:,0:128]
  conv<128,256,27,Z2c,Y2c,X2c, Z3c,Y3c,X3c, 2,1,0, 0,1,1>(stream, xc2, w_40, P(30), P(29), m3, nullptr, l3a, 256,0, 256,0, 0,0, 1,1);
  conv<256,256,27,Z3c,Y3c,X3c, Z3c,Y3c,X3c, 1,1,0, 1,1,1>(stream, l3a, w_41, P(33), P(32), m3, nullptr, l3b, 256,0, 256,0, 0,0, 1,1);
  conv<256,256,27,Z3c,Y3c,X3c, Z3c,Y3c,X3c, 1,1,0, 1,1,1>(stream, l3b, w_42, P(36), P(35), m3, nullptr, xc4, 256,0, 512,0, 0,0, 1,1); // c4 -> xc4[:,0:256]

  // decoder level 3
  conv<256,256,27,Z3c,Y3c,X3c, Z3c,Y3c,X3c, 1,1,0, 1,1,1>(stream, xc4, w_t4a, P(89), P(87), m3, nullptr, h3b, 512,0, 256,0, 0,0, 1,1);
  conv<256,256,27,Z3c,Y3c,X3c, Z3c,Y3c,X3c, 1,1,0, 1,1,1>(stream, h3b, w_t4b, P(90), P(88), m3, xc4, xc4, 256,0, 512,256, 512,0, 1,1); // xt -> xc4[:,256:512]
  conv<512,256,27,Z3c,Y3c,X3c, Z3c,Y3c,X3c, 1,1,0, 1,1,1>(stream, xc4, w_m4, P(78), P(77), m3, nullptr, xm4, 512,0, 256,0, 0,0, 1,1);
  pair_sum_k<<<cdiv(V3c * 256, 256), 256, 0, stream>>>(xm4, xc4, t4, V3c, 256);
  conv<256,128,27,Z3c,Y3c,X3c, Z2c,Y2c,X2c, 1,2,1, 0,1,1>(stream, t4, w_i4, P(58), P(57), m2, nullptr, xu4, 256,0, 128,0, 0,0, 1,1); // transposed

  // decoder level 2
  conv<128,128,27,Z2c,Y2c,X2c, Z2c,Y2c,X2c, 1,1,0, 1,1,1>(stream, xu4, w_t3a, P(83), P(81), m2, nullptr, h2b, 128,0, 128,0, 0,0, 1,1);
  conv<128,128,27,Z2c,Y2c,X2c, Z2c,Y2c,X2c, 1,1,0, 1,1,1>(stream, h2b, w_t3b, P(84), P(82), m2, xu4, xc2, 128,0, 256,128, 128,0, 1,1); // xt -> xc2[:,128:256]
  conv<256,128,27,Z2c,Y2c,X2c, Z2c,Y2c,X2c, 1,1,0, 1,1,1>(stream, xc2, w_m3, P(75), P(74), m2, nullptr, xm3, 256,0, 128,0, 0,0, 1,1);
  pair_sum_k<<<cdiv(V2c * 128, 256), 256, 0, stream>>>(xm3, xc2, t3, V2c, 128);
  conv<128,64,27, Z2c,Y2c,X2c, Z1c,Y1c,X1c, 1,2,1, 1,1,1>(stream, t3, w_i3, P(55), P(54), m1, nullptr, l1a, 128,0, 64,0, 0,0, 1,1); // xu3 -> l1a

  // embeddings (1x1 GEMMs)
  centers_k<<<cdiv(V1c, 256), 256, 0, stream>>>(cent, out_pc, V1c);
  conv<3,64,1,  V1c,1,1, V1c,1,1, 1,1,0, 0,0,0>(stream, cent, w_pe1, nullptr, P(70), nullptr, nullptr, peh, 3,0, 64,0, 0,0, 2,0);
  conv<64,64,1, V1c,1,1, V1c,1,1, 1,1,0, 0,0,0>(stream, peh, w_pe2, nullptr, P(71), nullptr, nullptr, pe, 64,0, 64,0, 0,0, 0,0);
  conv<3,64,1,  NHc,1,1, NHc,1,1, 1,1,0, 0,0,0>(stream, hpos, w_pe1, nullptr, P(70), nullptr, nullptr, hh, 3,0, 64,0, 0,0, 2,0);
  conv<64,64,1, NHc,1,1, NHc,1,1, 1,1,0, 0,0,0>(stream, hh, w_pe2, nullptr, P(71), nullptr, nullptr, hm2, 64,0, 64,0, 0,0, 0,0);
  hist_scatter_k<<<cdiv(NH * 64, 256), 256, 0, stream>>>(hfeat, hcrd, hmask, hm2, P(72), hd, mh, NH);
  fuse_k<<<cdiv(V1c * 64, 256), 256, 0, stream>>>(l1a, pe, P(72) + 64, m1, hd, mh, l1b, mu, V1c);

  // fusion + out convs (ping-pong l1a/l1b)
  conv<64,64,27, Z1c,Y1c,X1c, Z1c,Y1c,X1c, 1,1,0, 1,1,1>(stream, l1b, w_f0, P(42), P(41), mu, nullptr, l1a, 64,0, 64,0, 0,0, 1,1);
  conv<64,64,27, Z1c,Y1c,X1c, Z1c,Y1c,X1c, 1,1,0, 1,1,1>(stream, l1a, w_f1, P(45), P(44), mu, nullptr, l1b, 64,0, 64,0, 0,0, 1,1);
  conv<64,64,27, Z1c,Y1c,X1c, Z1c,Y1c,X1c, 1,1,0, 1,1,1>(stream, l1b, w_f2, P(48), P(47), mu, nullptr, out_sv, 64,0, 64,0, 0,0, 1,1); // save_features
  conv<64,64,27, Z1c,Y1c,X1c, Z1c,Y1c,X1c, 1,1,0, 1,1,1>(stream, out_sv, w_o0, P(61), P(60), mu, nullptr, l1a, 64,0, 64,0, 0,0, 1,1);
  conv<64,64,27, Z1c,Y1c,X1c, Z1c,Y1c,X1c, 1,1,0, 1,1,1>(stream, l1a, w_o1, P(64), P(63), mu, nullptr, l1b, 64,0, 64,0, 0,0, 1,1);
  conv<64,64,27, Z1c,Y1c,X1c, Z1c,Y1c,X1c, 1,1,0, 1,1,1>(stream, l1b, w_o2, P(67), P(66), mu, nullptr, l1a, 64,0, 64,0, 0,0, 1,1); // feats

  // gaussian head
  conv<64,64,1, V1c,1,1, V1c,1,1, 1,1,0, 0,0,0>(stream, l1a, w_g1, nullptr, P(51), nullptr, nullptr, l1b, 64,0, 64,0, 0,0, 2,0);
  conv<64,30,1, V1c,1,1, V1c,1,1, 1,1,0, 0,0,0>(stream, l1b, w_g2, nullptr, P(52), mu, nullptr, out_g, 64,0, 30,0, 0,0, 0,1);
  copy_mask_k<<<cdiv(V1c, 256), 256, 0, stream>>>(mu, out_mk, V1c);
#undef P
}
